// quantumAttentionBlock_74663711474326
// MI455X (gfx1250) — matrix operand order repaired and checked by reading, not compiled, not run on hardware
//
#include <hip/hip_runtime.h>
#include <hip/hip_bf16.h>
#include <math.h>

// ---------------------------------------------------------------------------
// MI455X (gfx1250): all GEMMs through v_wmma_f32_16x16x32_bf16, wave32.
// All WMMA operands are loaded as contiguous 16B chunks (weights stored
// transposed [N][K]); fragments assembled via dword bit-casts (no b16 moves).
// ---------------------------------------------------------------------------

typedef __bf16 bf16_t;
typedef __attribute__((ext_vector_type(16))) __bf16 v16bf;
typedef __attribute__((ext_vector_type(8)))  float  v8f;
typedef __attribute__((ext_vector_type(4)))  int    v4i_t;
typedef __attribute__((ext_vector_type(8)))  int    v8i_t;

#define B_   128
#define N_   197
#define D_   384
#define H_   1536
#define NP_  208          // N padded to multiple of 16
#define NT_  13           // NP_/16
#define NPK_ 224          // attn K padded to multiple of 32

// ---------------- WMMA helpers (ISA 7.12.2 wave32 layouts) ------------------

__device__ __forceinline__ v8f wmma_bf16(v16bf a, v16bf b, v8f c) {
    return __builtin_amdgcn_wmma_f32_16x16x32_bf16(
        false, a, false, b, (short)0, c, false, false);
}

// Assemble a 16-element bf16 fragment from two contiguous 16B chunks.
__device__ __forceinline__ v16bf frag_from2(const bf16_t* p0, const bf16_t* p1) {
    v4i_t lo = *(const v4i_t*)p0;
    v4i_t hi = *(const v4i_t*)p1;
    union { v8i_t i; v16bf b; } u;
    u.i[0] = lo[0]; u.i[1] = lo[1]; u.i[2] = lo[2]; u.i[3] = lo[3];
    u.i[4] = hi[0]; u.i[5] = hi[1]; u.i[6] = hi[2]; u.i[7] = hi[3];
    return u.b;
}

// A fragment 16x32 from row-major matrix (lda in elements, 16B aligned rows).
// lane<16: row=lane, K = {0..7, 16..23} ; lane>=16: row=lane-16, K = {8..15, 24..31}
__device__ __forceinline__ v16bf load_a_rm(const bf16_t* __restrict__ A,
                                           int lda, int row0, int k0) {
    int lane = threadIdx.x & 31;
    const bf16_t* p = A + (size_t)(row0 + (lane & 15)) * lda + k0 + ((lane >> 4) << 3);
    return frag_from2(p, p + 16);
}

// B fragment 32x16 where logical B[k][n] = M[n][k] (M row-major [N][K]).
// lane<16: col=lane, K = {0..7, 16..23} ; lane>=16: col=lane-16, K = {8..15, 24..31}
// -> two b128 loads per lane.
__device__ __forceinline__ v16bf load_bt_rm(const bf16_t* __restrict__ M,
                                            int ldm, int k0, int n0) {
    int lane = threadIdx.x & 31;
    const bf16_t* p = M + (size_t)(n0 + (lane & 15)) * ldm + k0 + ((lane >> 4) << 3);
    return frag_from2(p, p + 16);
}

__device__ __forceinline__ float gelu_tanh(float x) {
    float x3 = x * x * x;
    return 0.5f * x * (1.0f + tanhf(0.7978845608028654f * (x + 0.044715f * x3)));
}

// ---------------- generic wave-tiled WMMA GEMM (16x64 per wave) -------------
// B operand is given TRANSPOSED: Bt is [Nn][K] row-major, ldb = its row stride.
// MODE 0: out bf16 = acc                          (y = inp @ W)
// MODE 2: out bf16 = gelu(acc + bias[n])          (MLP fc1)
// MODE 3: out f32  = acc + bias[n] + res[m,n]     (MLP fc2 + residual)
// MODE 4: out bf16 = sqrt(acc^2+1e-8), stored transposed per batch:
//         Out[(b*D + n)*NPK + (m % NP)]           (U^T for the attn GEMM)
template <int MODE>
__global__ __launch_bounds__(256)
void gemm_wmma_bf16(const bf16_t* __restrict__ A, int lda,
                    const bf16_t* __restrict__ Bt, int ldb,
                    void* __restrict__ Out, int ldo,
                    int M, int Nn, int K,
                    const float* __restrict__ bias,
                    const float* __restrict__ res) {
    int wave   = (blockIdx.x * blockDim.x + threadIdx.x) >> 5;
    int ntiles = Nn >> 6;
    int mt = wave / ntiles;
    int nt = wave % ntiles;
    int m0 = mt << 4;
    int n0 = nt << 6;
    if (m0 >= M) return;

    v8f acc0 = {}, acc1 = {}, acc2 = {}, acc3 = {};
    for (int k0 = 0; k0 < K; k0 += 32) {
        if (k0 + 32 < K) {
            __builtin_prefetch(A  + (size_t)m0 * lda + k0 + 32, 0, 1);
            __builtin_prefetch(Bt + (size_t)n0 * ldb + k0 + 32, 0, 1);
        }
        v16bf a = load_a_rm(A, lda, m0, k0);
        acc0 = wmma_bf16(a, load_bt_rm(Bt, ldb, k0, n0     ), acc0);
        acc1 = wmma_bf16(a, load_bt_rm(Bt, ldb, k0, n0 + 16), acc1);
        acc2 = wmma_bf16(a, load_bt_rm(Bt, ldb, k0, n0 + 32), acc2);
        acc3 = wmma_bf16(a, load_bt_rm(Bt, ldb, k0, n0 + 48), acc3);
    }

    int lane = threadIdx.x & 31;
    int nl = lane & 15;
    int mh = (lane >> 4) << 3;
    int bb = m0 / NP_;             // only used by MODE 4
    int mm0 = m0 - bb * NP_;
#pragma unroll
    for (int j = 0; j < 8; ++j) {
        int m = m0 + mh + j;
        float v4[4] = {acc0[j], acc1[j], acc2[j], acc3[j]};
#pragma unroll
        for (int t = 0; t < 4; ++t) {
            int n = n0 + (t << 4) + nl;
            float v = v4[t];
            if (MODE == 0) {
                ((bf16_t*)Out)[(size_t)m * ldo + n] = (bf16_t)v;
            } else if (MODE == 2) {
                ((bf16_t*)Out)[(size_t)m * ldo + n] = (bf16_t)gelu_tanh(v + bias[n]);
            } else if (MODE == 3) {
                size_t idx = (size_t)m * ldo + n;
                ((float*)Out)[idx] = v + bias[n] + res[idx];
            } else { // MODE 4: transposed per-batch store of sqrt(v^2+eps)
                ((bf16_t*)Out)[((size_t)bb * D_ + n) * NPK_ + mm0 + mh + j] =
                    (bf16_t)sqrtf(v * v + 1e-8f);
            }
        }
    }
}

// ---------------- fused attention: scores -> softmax -> attn@U --------------
// 4 waves per block, each wave owns one (b, 16-row) query tile.
// Ut is U transposed per batch: [B][D][NPK] bf16 (pad K region pre-zeroed).
__global__ __launch_bounds__(128)
void attn_wmma(const bf16_t* __restrict__ y,     // [B, NP, D] bf16
               const bf16_t* __restrict__ inp,   // [B, NP, D] bf16
               const bf16_t* __restrict__ Ut,    // [B, D, NPK] bf16
               const float*  __restrict__ x,     // [B, N, D]  f32
               float* __restrict__ xr) {         // [B, N, D]  f32 out (x + t)
    __shared__ float sc[4][16][NP_];              // 53 KB (<64 KB)

    int wv   = threadIdx.x >> 5;
    int lane = threadIdx.x & 31;
    int w    = blockIdx.x * 4 + wv;
    int b    = w / NT_;
    int mt   = w % NT_;
    int m0   = mt << 4;

    const bf16_t* yb  = y   + (size_t)b * NP_ * D_;
    const bf16_t* ib  = inp + (size_t)b * NP_ * D_;
    const bf16_t* utb = Ut  + (size_t)b * D_ * NPK_;

    int nl = lane & 15;
    int mh = (lane >> 4) << 3;

    // ---- phase 1: register-resident query tile, scores for all 13 col tiles
    v16bf aY[D_ / 32];
#pragma unroll
    for (int kc = 0; kc < D_ / 32; ++kc) aY[kc] = load_a_rm(yb, D_, m0, kc * 32);

    for (int ct = 0; ct < NT_; ++ct) {
        v8f acc = {};
#pragma unroll
        for (int kc = 0; kc < D_ / 32; ++kc) {
            // B[k=e][n=m] = inp[m][e] -> transposed-layout load, contiguous K
            acc = wmma_bf16(aY[kc], load_bt_rm(ib, D_, kc * 32, ct << 4), acc);
        }
        int col = (ct << 4) + nl;
#pragma unroll
        for (int j = 0; j < 8; ++j) {
            float s = acc[j];
            s = sqrtf(s * s + 1e-8f);
            if (col >= N_) s = -1e30f;
            sc[wv][mh + j][col] = s;
        }
    }
    __syncthreads();

    // ---- phase 2: softmax per row, convert in place f32 -> bf16 (K padded)
    if (lane < 16) {
        float* r = &sc[wv][lane][0];
        float mx = -1e30f;
        for (int c = 0; c < NP_; ++c) mx = fmaxf(mx, r[c]);
        float s = 0.0f;
        for (int c = 0; c < NP_; ++c) s += __expf(r[c] - mx);
        float inv = 1.0f / s;
        bf16_t* ar = (bf16_t*)r;                  // bf16 write at 2c < f32 read at 4c'
        for (int c = 0; c < NP_; ++c) ar[c] = (bf16_t)(__expf(r[c] - mx) * inv);
        for (int c = NP_; c < NPK_; ++c) ar[c] = (bf16_t)0.0f;
    }
    __syncthreads();

    // ---- phase 3: t = attn @ U ; out = x + t
    // A fragments once from LDS (ds_load), reused across all 24 output tiles.
    const bf16_t* attnA = (const bf16_t*)&sc[wv][0][0];
    const int ldattn = NP_ * 2;                   // row stride in bf16 elements
    v16bf aA[NPK_ / 32];
#pragma unroll
    for (int kc = 0; kc < NPK_ / 32; ++kc)
        aA[kc] = load_a_rm(attnA, ldattn, 0, kc * 32);

    for (int nt2 = 0; nt2 < D_ / 16; ++nt2) {
        v8f acc = {};
#pragma unroll
        for (int kc = 0; kc < NPK_ / 32; ++kc) {
            // B[k=m][n=d] = Ut[d][m] -> contiguous K loads, pad K is zero
            acc = wmma_bf16(aA[kc], load_bt_rm(utb, NPK_, kc * 32, nt2 << 4), acc);
        }
#pragma unroll
        for (int j = 0; j < 8; ++j) {
            int m = m0 + mh + j;
            if (m < N_) {
                size_t idx = ((size_t)b * N_ + m) * D_ + (nt2 << 4) + nl;
                xr[idx] = x[idx] + acc[j];
            }
        }
    }
}

// ---------------- LayerNorm / normalization / utility kernels ---------------

__device__ __forceinline__ float block_sum128(float v, float* red) {
    int t = threadIdx.x;
    red[t] = v; __syncthreads();
    for (int s = 64; s > 0; s >>= 1) {
        if (t < s) red[t] += red[t + s];
        __syncthreads();
    }
    float r = red[0]; __syncthreads();
    return r;
}

__global__ __launch_bounds__(128)
void ln1_reduce(const float* __restrict__ x, const float* __restrict__ g,
                const float* __restrict__ beta, float* __restrict__ lnf,
                float* __restrict__ colsum) {
    __shared__ float red[128];
    int row = blockIdx.x;                    // b*N + n
    int b   = row / N_;
    const float* xrow = x + (size_t)row * D_;

    float s = 0.0f;
    for (int i = threadIdx.x; i < D_; i += 128) s += xrow[i];
    float mean = block_sum128(s, red) / (float)D_;

    float v = 0.0f;
    for (int i = threadIdx.x; i < D_; i += 128) {
        float d = xrow[i] - mean; v += d * d;
    }
    float var  = block_sum128(v, red) / (float)D_;
    float rstd = rsqrtf(var + 1e-5f);

    for (int i = threadIdx.x; i < D_; i += 128) {
        float ln = (xrow[i] - mean) * rstd * g[i] + beta[i];
        lnf[(size_t)row * D_ + i] = ln;
        atomicAdd(&colsum[b * D_ + i], ln * ln);
    }
}

__global__ __launch_bounds__(128)
void norm_cols(const float* __restrict__ lnf, const float* __restrict__ colsum,
               bf16_t* __restrict__ inp) {
    int row = blockIdx.x;                    // b*NP + n
    int b = row / NP_;
    int n = row % NP_;
    for (int i = threadIdx.x; i < D_; i += 128) {
        bf16_t o = (bf16_t)0.0f;
        if (n < N_) {
            float rinv = rsqrtf(colsum[b * D_ + i] + (float)N_ * 1e-4f + 1e-8f);
            o = (bf16_t)(lnf[((size_t)b * N_ + n) * D_ + i] * rinv);
        }
        inp[(size_t)row * D_ + i] = o;
    }
}

__global__ __launch_bounds__(128)
void ln2_to_bf16(const float* __restrict__ xr, const float* __restrict__ g,
                 const float* __restrict__ beta, bf16_t* __restrict__ hn) {
    __shared__ float red[128];
    int row = blockIdx.x;
    const float* xrow = xr + (size_t)row * D_;

    float s = 0.0f;
    for (int i = threadIdx.x; i < D_; i += 128) s += xrow[i];
    float mean = block_sum128(s, red) / (float)D_;

    float v = 0.0f;
    for (int i = threadIdx.x; i < D_; i += 128) {
        float d = xrow[i] - mean; v += d * d;
    }
    float var  = block_sum128(v, red) / (float)D_;
    float rstd = rsqrtf(var + 1e-5f);

    for (int i = threadIdx.x; i < D_; i += 128) {
        hn[(size_t)row * D_ + i] =
            (bf16_t)((xrow[i] - mean) * rstd * g[i] + beta[i]);
    }
}

__global__ void zero_f32(float* __restrict__ p, int n) {
    int i = blockIdx.x * 256 + threadIdx.x;
    if (i < n) p[i] = 0.0f;
}

__global__ void zero_bf16(bf16_t* __restrict__ p, int n) {
    int i = blockIdx.x * 256 + threadIdx.x;
    if (i < n) p[i] = (bf16_t)0.0f;
}

__global__ void cvt_bf16(const float* __restrict__ src, bf16_t* __restrict__ dst, int n) {
    int i = blockIdx.x * 256 + threadIdx.x;
    if (i < n) dst[i] = (bf16_t)src[i];
}

// dst[c*rows + r] = src[r*cols + c]  (store weights as [N][K])
__global__ void cvt_bf16_T(const float* __restrict__ src, bf16_t* __restrict__ dst,
                           int rows, int cols) {
    int i = blockIdx.x * 256 + threadIdx.x;
    if (i < rows * cols) {
        int r = i / cols, c = i % cols;
        dst[(size_t)c * rows + r] = (bf16_t)src[(size_t)r * cols + c];
    }
}

// ---------------------------------------------------------------------------

extern "C" void kernel_launch(void* const* d_in, const int* in_sizes, int n_in,
                              void* d_out, int out_size, void* d_ws, size_t ws_size,
                              hipStream_t stream) {
    const float* x     = (const float*)d_in[0];
    const float* ln1_g = (const float*)d_in[1];
    const float* ln1_b = (const float*)d_in[2];
    const float* V     = (const float*)d_in[3];
    const float* W     = (const float*)d_in[4];
    const float* ln2_g = (const float*)d_in[5];
    const float* ln2_b = (const float*)d_in[6];
    const float* W1    = (const float*)d_in[7];
    const float* b1    = (const float*)d_in[8];
    const float* W2    = (const float*)d_in[9];
    const float* b2    = (const float*)d_in[10];

    char*  base = (char*)d_ws;
    size_t off  = 0;
    auto carve = [&](size_t bytes) {
        char* p = base + off;
        off += (bytes + 255) & ~(size_t)255;
        return p;
    };
    float*  xr     = (float*) carve((size_t)B_ * N_ * D_ * 4);    // x + t residual
    float*  colsum = (float*) carve((size_t)B_ * D_ * 4);
    bf16_t* inp_bf = (bf16_t*)carve((size_t)B_ * NP_ * D_ * 2);
    bf16_t* y_bf   = (bf16_t*)carve((size_t)B_ * NP_ * D_ * 2);
    bf16_t* ut_bf  = (bf16_t*)carve((size_t)B_ * D_ * NPK_ * 2);  // U^T, K padded
    bf16_t* hn_bf  = (bf16_t*)carve((size_t)B_ * N_ * D_ * 2);
    bf16_t* g_bf   = (bf16_t*)carve((size_t)B_ * N_ * H_ * 2);    // MLP hidden
    bf16_t* V_bf   = (bf16_t*)carve((size_t)D_ * D_ * 2);         // already [n=d][k=e]
    bf16_t* Wt_bf  = (bf16_t*)carve((size_t)D_ * D_ * 2);         // [N][K]
    bf16_t* W1t_bf = (bf16_t*)carve((size_t)H_ * D_ * 2);         // [H][D]
    bf16_t* W2t_bf = (bf16_t*)carve((size_t)D_ * H_ * 2);         // [D][H]
    float*  lnf    = (float*)g_bf;   // LN1 f32 output, dead before g_bf is live

    // 1) weights -> bf16. V is already B-friendly ([n=d][k=e]); others transpose.
    cvt_bf16  <<<(D_ * D_ + 255) / 256, 256, 0, stream>>>(V, V_bf, D_ * D_);
    cvt_bf16_T<<<(D_ * D_ + 255) / 256, 256, 0, stream>>>(W, Wt_bf, D_, D_);
    cvt_bf16_T<<<(D_ * H_ + 255) / 256, 256, 0, stream>>>(W1, W1t_bf, D_, H_);
    cvt_bf16_T<<<(H_ * D_ + 255) / 256, 256, 0, stream>>>(W2, W2t_bf, H_, D_);

    // 2) LN1 + patch-axis column norms
    zero_f32<<<(B_ * D_ + 255) / 256, 256, 0, stream>>>(colsum, B_ * D_);
    ln1_reduce<<<B_ * N_, 128, 0, stream>>>(x, ln1_g, ln1_b, lnf, colsum);
    norm_cols <<<B_ * NP_, 128, 0, stream>>>(lnf, colsum, inp_bf);

    // 3) projections: y = inp @ W ; Ut = sqrt((inp @ V^T)^2 + 1e-8) transposed
    zero_bf16<<<((int)((size_t)B_ * D_ * NPK_) + 255) / 256, 256, 0, stream>>>(
        ut_bf, (int)((size_t)B_ * D_ * NPK_));
    {
        int waves = (B_ * NP_ / 16) * (D_ / 64);   // 9984
        gemm_wmma_bf16<0><<<waves / 8, 256, 0, stream>>>(
            inp_bf, D_, Wt_bf, D_, y_bf, D_, B_ * NP_, D_, D_, nullptr, nullptr);
        gemm_wmma_bf16<4><<<waves / 8, 256, 0, stream>>>(
            inp_bf, D_, V_bf, D_, ut_bf, 0, B_ * NP_, D_, D_, nullptr, nullptr);
    }

    // 4) fused attention -> xr = x + attn @ U
    attn_wmma<<<B_ * NT_ / 4, 128, 0, stream>>>(y_bf, inp_bf, ut_bf, x, xr);

    // 5) LN2 -> bf16, then MLP (fc1 + gelu, fc2 + bias + residual)
    ln2_to_bf16<<<B_ * N_, 128, 0, stream>>>(xr, ln2_g, ln2_b, hn_bf);
    {
        int waves1 = (B_ * N_ / 16) * (H_ / 64);   // 37824
        gemm_wmma_bf16<2><<<waves1 / 8, 256, 0, stream>>>(
            hn_bf, D_, W1t_bf, D_, g_bf, H_, B_ * N_, H_, D_, b1, nullptr);
        int waves2 = (B_ * N_ / 16) * (D_ / 64);   // 9456
        gemm_wmma_bf16<3><<<waves2 / 8, 256, 0, stream>>>(
            g_bf, H_, W2t_bf, H_, d_out, D_, B_ * N_, D_, H_, b2, xr);
    }
}
